// OptimizedMambaGNN_3728031613172
// MI455X (gfx1250) — matrix operand order repaired and checked by reading, not compiled, not run on hardware
//
#include <hip/hip_runtime.h>
#include <hip/hip_bf16.h>
#include <math.h>

// CDNA5 (gfx1250) fused MambaGNN forward. wave32, WMMA 16x16x32 f16.
// Weights are pre-swizzled once per launch into WMMA B-fragment order (f16, in d_ws);
// activations feeding WMMA are mirrored once as f16 in LDS (A-fragment friendly).
// GEMM epilogues are branch-free: pad rows/cols are written but never read.
typedef __attribute__((ext_vector_type(16))) _Float16 v16h;
typedef __attribute__((ext_vector_type(8)))  _Float16 v8h;
typedef __attribute__((ext_vector_type(8)))  float    v8f;

#define NTH   128   // 4 wave32s per block, one block per sample
#define NWAVE 4

#define NP    14    // patches
#define D     64
#define LDA   68    // padded f32 row stride
#define DI    128
#define LDI   132
#define DS16  16
#define LDXD  48    // xdbl stride = padded N of xproj (cols 36..47 written, never read)
#define LDXZ  260
#define KNN   9
#define NCOUT 256
#define LDH   128   // f16 mirror stride (halves)

// ---- f32 shared memory layout (floats) ----
#define OFF_H    0        // [16][68]  mamba stream h
#define OFF_HG   1088     // [16][68]  gat stream
#define OFF_TMP  2176     // [16][68]  Wh / hf_pre / k-proj
#define OFF_XZ   3264     // [16][260] xbuf / xz
#define OFF_U    7424     // [16][132] conv-silu u / hf(ld 68)
#define OFF_XD   9536     // [16][48]  xdbl / GAT src-dst
#define OFF_DT   10304    // [16][132] dt / combmean / h1
#define OFF_MASK 12416    // [16][16]
#define OFF_RED  12672    // [64]
#define OFF_V    12736    // [512] small vectors
#define SM_TOTAL 13248    // 51.75 KB f32  (+4 KB f16 mirror)

#define V_CW  0
#define V_HM  64
#define V_Q   128
#define V_LG  192
#define V_AW  208
#define V_HGL 224

// ---- prepped-weight offsets in d_ws (halves) ----
#define WOFF_M0_IN    0        // 64x256
#define WOFF_M0_XP    16384    // 128x48 (36 padded)
#define WOFF_M0_OUT   22528    // 128x64
#define WOFF_M1_IN    30720
#define WOFF_M1_XP    47104
#define WOFF_M1_OUT   53248
#define WOFF_G0_W     61440    // 64x64
#define WOFF_G0_OUT   65536
#define WOFF_G1_W     69632
#define WOFF_G1_OUT   73728
#define WOFF_FUS      77824    // 128x64
#define WOFF_K        86016    // 64x64
#define WS_HALVES     90112    // + 64 f32 zeros after => d_ws needs 180480 B

struct MambaP {
  const float *ln_g, *ln_b, *in_b, *conv_w, *conv_b,
              *dt_w, *dt_b, *A_log, *Dp, *out_b, *xproj_b /*zeros*/;
  const _Float16 *in_wh, *xproj_wh, *out_wh;   // prepped B-fragment order
};
struct GatP {
  const float *W_b, *a_src, *a_dst, *out_b, *ln_g, *ln_b;
  const _Float16 *W_wh, *out_wh;
};
struct Params {
  const float *x;
  const float *patch_w, *patch_b, *pos, *in_ln_g, *in_ln_b;
  MambaP mamba[2];
  GatP   gat[2];
  const float *ca_w, *ca_b, *fus_b, *fus_ln_g, *fus_ln_b;
  const float *q_w, *q_b, *k_b;
  const float *c1_w, *c1_b, *c_ln_g, *c_ln_b, *c2_w, *c2_b;
  const _Float16 *fus_wh, *k_wh;
  float *out;
};

__device__ __forceinline__ float siluf(float x)  { return x / (1.f + __expf(-x)); }
__device__ __forceinline__ float sigf(float x)   { return 1.f / (1.f + __expf(-x)); }
__device__ __forceinline__ float geluf(float x)  { return 0.5f * x * (1.f + erff(x * 0.70710678118654752f)); }
__device__ __forceinline__ float softplusf(float x) { return (x > 20.f) ? x : log1pf(__expf(x)); }

// ============ weight pre-swizzle: [K][Nsrc] f32 -> B-fragment-ordered f16 ============
// element i = ((nt*(K/32)+kk)*32 + lane)*16 + e  maps to  k=(kk<<5)+((e>>3)<<4)+((lane>>4)<<3)+(e&7),
// n=(nt<<4)+(lane&15)   (CDNA5 ISA 7.12.2 16-bit B 32x16 per-lane layout)
struct WDesc { const float* src; int K, Nsrc, Npad, dstoff; };
struct PrepP { WDesc w[12]; _Float16* dst; float* zeros; };

__global__ void __launch_bounds__(256)
prep_weights_kernel(PrepP P)
{
  if (blockIdx.x == 12) {               // zeros bias block
    for (int i = threadIdx.x; i < 64; i += 256) P.zeros[i] = 0.f;
    return;
  }
  const WDesc wd = P.w[blockIdx.x];
  const int total = wd.K * wd.Npad;
  const int kst = wd.K >> 5;
  for (int i = threadIdx.x; i < total; i += 256) {
    const int e    = i & 15;
    const int lane = (i >> 4) & 31;
    const int blk  = i >> 9;
    const int kk   = blk % kst;
    const int nt   = blk / kst;
    const int k = (kk << 5) + ((e >> 3) << 4) + ((lane >> 4) << 3) + (e & 7);
    const int n = (nt << 4) + (lane & 15);
    const float v = (n < wd.Nsrc) ? wd.src[k * wd.Nsrc + n] : 0.f;
    P.dst[wd.dstoff + i] = (_Float16)v;
  }
}

// ============ WMMA GEMM: D = A(16xK f16 LDS) x B(prepped f16) + bias (+residual) ====
// A frag: lane l row m=l&15, halves = two contiguous 8-half (16B) LDS chunks at
//         k0+((l>>4)<<3) and k0+16+((l>>4)<<3)  -> ds_load_b128 x2 (reloaded per N tile)
// B frag: one contiguous 32B global load per lane from the prepped stream
// Epilogue: branch-free; bias must be non-null (use zeros block); all 16 rows stored
// (pad rows of f32 C buffers are never read; WMMA A always comes from the f16 mirror).
__device__ __forceinline__ void wmma_gemm(
    const _Float16* __restrict__ Ah,          // LDS, stride LDH, rows 14/15 zero
    const _Float16* __restrict__ Wp,          // prepped global
    int K,
    const float* __restrict__ bias,
    float* __restrict__ C, int ldc, int ntiles,
    int lane, int wave, bool add)
{
  const int m   = lane & 15;
  const int ka  = (lane >> 4) << 3;
  const int kst = K >> 5;
  for (int nt = wave; nt < ntiles; nt += NWAVE) {
    v8f acc = {};
    for (int kk = 0; kk < kst; ++kk) {
      const int k0 = kk << 5;
      const v8h lo = *(const v8h*)(Ah + m * LDH + k0 + ka);
      const v8h hi = *(const v8h*)(Ah + m * LDH + k0 + 16 + ka);
      const v16h af = __builtin_shufflevector(lo, hi,
          0, 1, 2, 3, 4, 5, 6, 7, 8, 9, 10, 11, 12, 13, 14, 15);
      const v16h bf = *(const v16h*)(Wp + (((nt * kst + kk) << 5) + lane) * 16);
      acc = __builtin_amdgcn_wmma_f32_16x16x32_f16(false, af, false, bf,
                                                   (short)0, acc, false, false);
    }
    const int nb = (nt << 4) + (lane & 15);
    const int mb = (lane >> 4) << 3;
    const float bv = bias[nb];
#pragma unroll
    for (int r = 0; r < 8; ++r) {
      const float v = acc[r] + bv;
      if (add) C[(mb + r) * ldc + nb] += v; else C[(mb + r) * ldc + nb] = v;
    }
  }
}

// LayerNorm rows -> f16 mirror only (feeds WMMA A)
__device__ __forceinline__ void ln_rows64_h(const float* in, int ldin, _Float16* outh,
                                            const float* g, const float* bta, int tid)
{
  if (tid < NP) {
    const int r = tid;
    float mmean = 0.f;
    for (int d = 0; d < 64; ++d) mmean += in[r * ldin + d];
    mmean *= (1.f / 64.f);
    float var = 0.f;
    for (int d = 0; d < 64; ++d) { float t = in[r * ldin + d] - mmean; var += t * t; }
    const float inv = rsqrtf(var * (1.f / 64.f) + 1e-5f);
    for (int d = 0; d < 64; ++d)
      outh[r * LDH + d] = (_Float16)((in[r * ldin + d] - mmean) * inv * g[d] + bta[d]);
  }
}

__device__ void mamba_block(const MambaP mp, float* sH, float* sXZ, float* sU,
                            float* sXD, float* sDT, _Float16* aA,
                            int tid, int lane, int wave)
{
  ln_rows64_h(sH, LDA, aA, mp.ln_g, mp.ln_b, tid);
  __syncthreads();
  // xz = LN(h) @ in_w[64,256] + in_b   (16 N-tiles x 2 K-steps)
  wmma_gemm(aA, mp.in_wh, 64, mp.in_b, sXZ, LDXZ, 16, lane, wave, false);
  __syncthreads();
  // causal depthwise conv(k=4) + silu; dual write f32 + f16 mirror (A of xproj)
  for (int idx = tid; idx < NP * DI; idx += NTH) {
    const int n = idx >> 7, c = idx & 127;
    const float* w = mp.conv_w + c * 4;
    float acc = 0.f;
#pragma unroll
    for (int j = 0; j < 4; ++j) {
      const int s = n - 3 + j;
      if (s >= 0) acc += sXZ[s * LDXZ + c] * w[j];
    }
    const float u = siluf(acc + mp.conv_b[c]);
    sU[n * LDI + c] = u;
    aA[n * LDH + c] = (_Float16)u;
  }
  __syncthreads();
  // xdbl = u @ xproj_w[128,36->48]   (3 N-tiles x 4 K-steps, zeros bias)
  wmma_gemm(aA, mp.xproj_wh, 128, mp.xproj_b, sXD, LDXD, 3, lane, wave, false);
  __syncthreads();
  // dt = softplus(dt_raw @ dt_w[4,128] + dt_b)
  for (int idx = tid; idx < NP * DI; idx += NTH) {
    const int n = idx >> 7, c = idx & 127;
    float t = mp.dt_b[c];
#pragma unroll
    for (int r = 0; r < 4; ++r) t += sXD[n * LDXD + r] * mp.dt_w[r * 128 + c];
    sDT[n * LDI + c] = softplusf(t);
  }
  __syncthreads();
  // selective scan: one channel per thread, state in registers; y -> f16 mirror only
  if (tid < DI) {
    const int c = tid;
    float negA[DS16], hst[DS16];
#pragma unroll
    for (int s = 0; s < DS16; ++s) { negA[s] = -__expf(mp.A_log[c * DS16 + s]); hst[s] = 0.f; }
    const float Dpc = mp.Dp[c];
    for (int n = 0; n < NP; ++n) {
      const float dt = sDT[n * LDI + c];
      const float u  = sU[n * LDI + c];
      const float du = dt * u;
      float y = 0.f;
#pragma unroll
      for (int s = 0; s < DS16; ++s) {
        const float Bv = sXD[n * LDXD + 4 + s];
        const float Cv = sXD[n * LDXD + 20 + s];
        hst[s] = __expf(dt * negA[s]) * hst[s] + du * Bv;
        y += hst[s] * Cv;
      }
      const float z = sXZ[n * LDXZ + 128 + c];
      aA[n * LDH + c] = (_Float16)((y + Dpc * u) * siluf(z));
    }
  }
  __syncthreads();
  // h += y @ out_w[128,64] + out_b   (residual accumulate; pad rows harmless)
  wmma_gemm(aA, mp.out_wh, 128, mp.out_b, sH, LDA, 4, lane, wave, true);
  __syncthreads();
}

__device__ void gat_block(const GatP gp, float* sHG, float* sTMP, float* sXD,
                          float* sU, _Float16* aA, const float* sMASK,
                          int tid, int lane, int wave)
{
  // Wh = hg @ W_w + W_b   (A = HG f16 mirror)
  wmma_gemm(aA, gp.W_wh, 64, gp.W_b, sTMP, LDA, 4, lane, wave, false);
  __syncthreads();
  // per-head src/dst scores
  if (tid < NP * 8) {
    const int i = tid >> 3, hh = tid & 7;
    float s = 0.f, t = 0.f;
#pragma unroll
    for (int d = 0; d < 8; ++d) {
      const float w = sTMP[i * LDA + hh * 8 + d];
      s += w * gp.a_src[hh * 8 + d];
      t += w * gp.a_dst[hh * 8 + d];
    }
    sXD[i * 8 + hh] = s;
    sXD[128 + i * 8 + hh] = t;
  }
  __syncthreads();
  // masked softmax over neighbors + aggregate -> f16 mirror only (A of out-proj)
  if (tid < NP * 8) {
    const int i = tid >> 3, hh = tid & 7;
    const float si = sXD[i * 8 + hh];
    float ev[NP], mx = -1e30f;
    for (int j = 0; j < NP; ++j) {
      float e;
      if (sMASK[i * 16 + j] > 0.5f) {
        const float t = si + sXD[128 + j * 8 + hh];
        e = (t < 0.f) ? 0.2f * t : t;
      } else e = -1e9f;
      ev[j] = e; mx = fmaxf(mx, e);
    }
    float ssum = 0.f;
    for (int j = 0; j < NP; ++j) { ev[j] = __expf(ev[j] - mx); ssum += ev[j]; }
    const float inv = 1.f / ssum;
#pragma unroll
    for (int d = 0; d < 8; ++d) {
      float o = 0.f;
      for (int j = 0; j < NP; ++j) o += ev[j] * sTMP[j * LDA + hh * 8 + d];
      aA[i * LDH + hh * 8 + d] = (_Float16)(o * inv);
    }
  }
  __syncthreads();
  // out = att @ out_w + out_b
  wmma_gemm(aA, gp.out_wh, 64, gp.out_b, sU, LDA, 4, lane, wave, false);
  __syncthreads();
  // hg = LN(hg + out); refresh f16 mirror for next layer's Wh GEMM
  if (tid < NP) {
    const int r = tid;
    float mmean = 0.f;
    for (int d = 0; d < 64; ++d) mmean += sHG[r * LDA + d] + sU[r * LDA + d];
    mmean *= (1.f / 64.f);
    float var = 0.f;
    for (int d = 0; d < 64; ++d) { float t = sHG[r * LDA + d] + sU[r * LDA + d] - mmean; var += t * t; }
    const float inv = rsqrtf(var * (1.f / 64.f) + 1e-5f);
    for (int d = 0; d < 64; ++d) {
      const float v = (sHG[r * LDA + d] + sU[r * LDA + d] - mmean) * inv * gp.ln_g[d] + gp.ln_b[d];
      sHG[r * LDA + d] = v;
      aA[r * LDH + d] = (_Float16)v;
    }
  }
  __syncthreads();
}

__global__ void __launch_bounds__(NTH)
mambagnn_fused_kernel(Params P)
{
  __shared__ float sm[SM_TOTAL];
  alignas(32) __shared__ _Float16 aA[16 * LDH];   // WMMA A mirror, rows 14/15 kept zero
  float* sH    = sm + OFF_H;
  float* sHG   = sm + OFF_HG;
  float* sTMP  = sm + OFF_TMP;
  float* sXZ   = sm + OFF_XZ;
  float* sU    = sm + OFF_U;
  float* sXD   = sm + OFF_XD;
  float* sDT   = sm + OFF_DT;
  float* sMASK = sm + OFF_MASK;
  float* sRED  = sm + OFF_RED;
  float* sV    = sm + OFF_V;

  const int tid  = threadIdx.x;
  const int lane = tid & 31;
  const int wave = tid >> 5;
  const int b    = blockIdx.x;

  __builtin_prefetch(P.c2_w + tid * 512, 0, 0);  // warm L2 with largest shared weight

  for (int i = tid; i < SM_TOTAL; i += NTH) sm[i] = 0.f;
  for (int i = tid; i < 16 * LDH; i += NTH) aA[i] = (_Float16)0.f;
  __syncthreads();

  // ---- stage x sample + patch embed + LN ----
  const float* xg = P.x + (size_t)b * 700;
  for (int i = tid; i < 700; i += NTH) sXZ[i] = xg[i];
  __syncthreads();
  for (int idx = tid; idx < NP * D; idx += NTH) {
    const int n = idx >> 6, d = idx & 63;
    const float* w  = P.patch_w + d * 50;
    const float* xp = sXZ + n * 50;
    float s = 0.f;
#pragma unroll 10
    for (int k = 0; k < 50; ++k) s += xp[k] * w[k];
    sH[n * LDA + d] = s + P.patch_b[d] + P.pos[n * 64 + d];
  }
  __syncthreads();
  if (tid < NP) {  // LN (in_ln) in place on sH
    const int r = tid;
    float mmean = 0.f;
    for (int d = 0; d < 64; ++d) mmean += sH[r * LDA + d];
    mmean *= (1.f / 64.f);
    float var = 0.f;
    for (int d = 0; d < 64; ++d) { float t = sH[r * LDA + d] - mmean; var += t * t; }
    const float inv = rsqrtf(var * (1.f / 64.f) + 1e-5f);
    for (int d = 0; d < 64; ++d)
      sH[r * LDA + d] = (sH[r * LDA + d] - mmean) * inv * P.in_ln_g[d] + P.in_ln_b[d];
  }
  __syncthreads();

  // ---- 2x mamba ----
  for (int blk = 0; blk < 2; ++blk)
    mamba_block(P.mamba[blk], sH, sXZ, sU, sXD, sDT, aA, tid, lane, wave);

  // ---- kNN top-9 mask (cosine sim; adj>0 semantics) ----
  if (tid < NP) {
    float ss = 0.f;
    for (int d = 0; d < 64; ++d) { const float v = sH[tid * LDA + d]; ss += v * v; }
    sRED[tid] = 1.f / fmaxf(sqrtf(ss), 1e-12f);
  }
  __syncthreads();
  if (tid < NP) {
    const int r = tid;
    float sim[NP];
    for (int j = 0; j < NP; ++j) {
      float s = 0.f;
      for (int d = 0; d < 64; ++d) s += sH[r * LDA + d] * sH[j * LDA + d];
      sim[j] = s * sRED[r] * sRED[j];
    }
    for (int j = 0; j < 16; ++j) sMASK[r * 16 + j] = 0.f;
    unsigned used = 0;
    for (int t = 0; t < KNN; ++t) {
      int bi = 0; float bv = -1e30f;
      for (int j = 0; j < NP; ++j)
        if (!((used >> j) & 1u) && sim[j] > bv) { bv = sim[j]; bi = j; }
      used |= 1u << bi;
      if (bv > 0.f) sMASK[r * 16 + bi] = 1.f;
    }
  }
  // hg = h (f32 + f16 mirror for first GAT GEMM)
  for (int idx = tid; idx < NP * D; idx += NTH) {
    const int n = idx >> 6, d = idx & 63;
    const float v = sH[n * LDA + d];
    sHG[n * LDA + d] = v;
    aA[n * LDH + d] = (_Float16)v;
  }
  __syncthreads();

  // ---- 2x GAT ----
  for (int blk = 0; blk < 2; ++blk)
    gat_block(P.gat[blk], sHG, sTMP, sXD, sU, aA, sMASK, tid, lane, wave);

  // ---- fusion: channel attention + projection ----
  for (int c = tid; c < 2 * D; c += NTH) {           // mean over patches of comb
    float s = 0.f;
    for (int n = 0; n < NP; ++n)
      s += (c < D) ? sH[n * LDA + c] : sHG[n * LDA + (c - D)];
    sDT[c] = s * (1.f / (float)NP);
  }
  __syncthreads();
  for (int c = tid; c < D; c += NTH) {               // cw = sigmoid(cm @ ca_w + ca_b)
    float s = P.ca_b[c];
    for (int k = 0; k < 2 * D; ++k) s += sDT[k] * P.ca_w[k * D + c];
    sV[V_CW + c] = sigf(s);
  }
  for (int idx = tid; idx < 16 * 2 * D; idx += NTH) { // comb f16 mirror [16][128], pads zero
    const int n = idx >> 7, c = idx & 127;
    aA[idx] = (n < NP) ? (_Float16)((c < D) ? sH[n * LDA + c] : sHG[n * LDA + (c - D)])
                       : (_Float16)0.f;
  }
  __syncthreads();
  // hf_pre = comb @ fus_w[128,64] + fus_b
  wmma_gemm(aA, P.fus_wh, 128, P.fus_b, sTMP, LDA, 4, lane, wave, false);
  __syncthreads();
  if (tid < NP) {                                    // hf = gelu(LN(hf_pre)) * cw
    const int r = tid;
    float mmean = 0.f;
    for (int d = 0; d < 64; ++d) mmean += sTMP[r * LDA + d];
    mmean *= (1.f / 64.f);
    float var = 0.f;
    for (int d = 0; d < 64; ++d) { float t = sTMP[r * LDA + d] - mmean; var += t * t; }
    const float inv = rsqrtf(var * (1.f / 64.f) + 1e-5f);
    for (int d = 0; d < 64; ++d) {
      const float t = (sTMP[r * LDA + d] - mmean) * inv * P.fus_ln_g[d] + P.fus_ln_b[d];
      const float hf = geluf(t) * sV[V_CW + d];
      sU[r * LDA + d] = hf;                 // f32 for pooling
      aA[r * LDH + d] = (_Float16)hf;       // A of k-projection
    }
  }
  __syncthreads();

  // ---- attention pooling ----
  for (int d = tid; d < D; d += NTH) {
    float s = 0.f;
    for (int n = 0; n < NP; ++n) s += sU[n * LDA + d];
    sV[V_HM + d] = s * (1.f / (float)NP);
  }
  __syncthreads();
  for (int d = tid; d < D; d += NTH) {
    float s = P.q_b[d];
    for (int k = 0; k < D; ++k) s += sV[V_HM + k] * P.q_w[k * D + d];
    sV[V_Q + d] = s;
  }
  __syncthreads();
  wmma_gemm(aA, P.k_wh, 64, P.k_b, sTMP, LDA, 4, lane, wave, false); // k = hf @ k_w + k_b
  __syncthreads();
  if (tid < NP) {
    float s = 0.f;
    for (int d = 0; d < D; ++d) s += sV[V_Q + d] * sTMP[tid * LDA + d];
    sV[V_LG + tid] = s * 0.125f;   // 1/sqrt(64)
  }
  __syncthreads();
  if (tid == 0) {
    float mx = -1e30f;
    for (int n = 0; n < NP; ++n) mx = fmaxf(mx, sV[V_LG + n]);
    float ssum = 0.f;
    for (int n = 0; n < NP; ++n) { const float e = __expf(sV[V_LG + n] - mx); sV[V_AW + n] = e; ssum += e; }
    const float inv = 1.f / ssum;
    for (int n = 0; n < NP; ++n) sV[V_AW + n] *= inv;
  }
  __syncthreads();
  for (int d = tid; d < D; d += NTH) {
    float s = 0.f;
    for (int n = 0; n < NP; ++n) s += sV[V_AW + n] * sU[n * LDA + d];
    sV[V_HGL + d] = s;
  }
  __syncthreads();

  // ---- classifier ----
  for (int j = tid; j < 256; j += NTH) {
    float s = P.c1_b[j];
    for (int k = 0; k < D; ++k) s += sV[V_HGL + k] * P.c1_w[k * 256 + j];
    sDT[j] = s;
  }
  __syncthreads();
  if (tid == 0) {
    float mmean = 0.f;
    for (int j = 0; j < 256; ++j) mmean += sDT[j];
    mmean *= (1.f / 256.f);
    float var = 0.f;
    for (int j = 0; j < 256; ++j) { float t = sDT[j] - mmean; var += t * t; }
    sRED[0] = mmean;
    sRED[1] = rsqrtf(var * (1.f / 256.f) + 1e-5f);
  }
  __syncthreads();
  for (int j = tid; j < 256; j += NTH)
    sDT[j] = geluf((sDT[j] - sRED[0]) * sRED[1] * P.c_ln_g[j] + P.c_ln_b[j]);
  __syncthreads();
  float* og = P.out + (size_t)b * NCOUT;
  for (int j = tid; j < NCOUT; j += NTH) {
    float s = P.c2_b[j];
    for (int k = 0; k < 256; ++k) s += sDT[k] * P.c2_w[k * 256 + j];
    og[j] = s;
  }
}

extern "C" void kernel_launch(void* const* d_in, const int* in_sizes, int n_in,
                              void* d_out, int out_size, void* d_ws, size_t ws_size,
                              hipStream_t stream)
{
  (void)n_in; (void)out_size; (void)ws_size;  // needs >= 180480 B scratch
  // Inputs flattened in setup_inputs() insertion order:
  // x, patch_w, patch_b, pos, in_ln_g, in_ln_b,
  // 2x mamba{ln_g,ln_b,in_w,in_b,conv_w,conv_b,xproj_w,dt_w,dt_b,A_log,Dp,out_w,out_b},
  // 2x gat{W_w,W_b,a_src,a_dst,out_w,out_b,ln_g,ln_b},
  // ca_w, ca_b, fus_w, fus_b, fus_ln_g, fus_ln_b, q_w, q_b, k_w, k_b,
  // c1_w, c1_b, c_ln_g, c_ln_b, c2_w, c2_b
  int i = 0;
  auto nxt = [&]() { return (const float*)d_in[i++]; };
  _Float16* wsh   = (_Float16*)d_ws;
  float*    zeros = (float*)((char*)d_ws + (size_t)WS_HALVES * 2);

  Params P;
  PrepP  PP;
  PP.dst = wsh;
  PP.zeros = zeros;
  int wi = 0;
  auto addw = [&](const float* src, int K, int Nsrc, int Npad, int off) {
    PP.w[wi].src = src; PP.w[wi].K = K; PP.w[wi].Nsrc = Nsrc;
    PP.w[wi].Npad = Npad; PP.w[wi].dstoff = off; ++wi;
  };

  P.x = nxt();
  P.patch_w = nxt(); P.patch_b = nxt(); P.pos = nxt();
  P.in_ln_g = nxt(); P.in_ln_b = nxt();
  const int m_in[2]  = {WOFF_M0_IN,  WOFF_M1_IN};
  const int m_xp[2]  = {WOFF_M0_XP,  WOFF_M1_XP};
  const int m_out[2] = {WOFF_M0_OUT, WOFF_M1_OUT};
  for (int m = 0; m < 2; ++m) {
    MambaP& mp = P.mamba[m];
    mp.ln_g = nxt();   mp.ln_b = nxt();
    const float* in_w = nxt();   mp.in_b = nxt();
    mp.conv_w = nxt(); mp.conv_b = nxt();
    const float* xproj_w = nxt();
    mp.dt_w = nxt();   mp.dt_b = nxt();
    mp.A_log = nxt();  mp.Dp = nxt();
    const float* out_w = nxt();  mp.out_b = nxt();
    mp.xproj_b = zeros;
    addw(in_w,    64, 256, 256, m_in[m]);   mp.in_wh    = wsh + m_in[m];
    addw(xproj_w, 128, 36, 48,  m_xp[m]);   mp.xproj_wh = wsh + m_xp[m];
    addw(out_w,   128, 64, 64,  m_out[m]);  mp.out_wh   = wsh + m_out[m];
  }
  const int g_w[2]   = {WOFF_G0_W,   WOFF_G1_W};
  const int g_out[2] = {WOFF_G0_OUT, WOFF_G1_OUT};
  for (int g = 0; g < 2; ++g) {
    GatP& gp = P.gat[g];
    const float* W_w = nxt();   gp.W_b = nxt();
    gp.a_src = nxt(); gp.a_dst = nxt();
    const float* out_w = nxt(); gp.out_b = nxt();
    gp.ln_g = nxt();  gp.ln_b = nxt();
    addw(W_w,   64, 64, 64, g_w[g]);    gp.W_wh   = wsh + g_w[g];
    addw(out_w, 64, 64, 64, g_out[g]);  gp.out_wh = wsh + g_out[g];
  }
  P.ca_w = nxt();  P.ca_b = nxt();
  const float* fus_w = nxt(); P.fus_b = nxt();
  P.fus_ln_g = nxt(); P.fus_ln_b = nxt();
  P.q_w = nxt();  P.q_b = nxt();
  const float* k_w = nxt();  P.k_b = nxt();
  P.c1_w = nxt(); P.c1_b = nxt();
  P.c_ln_g = nxt(); P.c_ln_b = nxt();
  P.c2_w = nxt(); P.c2_b = nxt();
  addw(fus_w, 128, 64, 64, WOFF_FUS);  P.fus_wh = wsh + WOFF_FUS;
  addw(k_w,    64, 64, 64, WOFF_K);    P.k_wh   = wsh + WOFF_K;
  P.out = (float*)d_out;

  // 1) pre-swizzle weights into WMMA B-fragment order + zeros bias (deterministic)
  prep_weights_kernel<<<13, 256, 0, stream>>>(PP);
  // 2) fused forward, one workgroup (4 wave32s) per sample
  const int B = in_sizes[0] / 700;   // 4096
  mambagnn_fused_kernel<<<B, NTH, 0, stream>>>(P);
}
